// Sae_14250701488368
// MI455X (gfx1250) — matrix operand order repaired and checked by reading, not compiled, not run on hardware
//
#include <hip/hip_runtime.h>
#include <hip/hip_bf16.h>

#define TOK   2048
#define DIN   2048
#define DSAE  65536
#define KTOP  32

#define BM 128
#define BN 128
#define BK 32
#define NROWG 8   // row-groups for column stats partials

typedef __attribute__((ext_vector_type(16))) __bf16 v16bf;
typedef __attribute__((ext_vector_type(8)))  __bf16 v8bf;
typedef __attribute__((ext_vector_type(8)))  float  v8f;
typedef __attribute__((ext_vector_type(4)))  float  v4f;

static __device__ __forceinline__ __bf16 f2bf(float f) { return (__bf16)f; }

// ---------------------------------------------------------------------------
// Kernel A: feats = relu((x - b_dec) @ W_enc + b_enc), bf16 WMMA 16x16x32.
// grid = (TOK/BM, DSAE/BN): M fastest so the 16 M-blocks sharing a W_enc
// column panel run together and hit it in L2 (W_enc streamed from HBM once).
// Register-staged global->LDS pipeline hides HBM latency behind the WMMAs.
// ---------------------------------------------------------------------------
__global__ __launch_bounds__(256) void sae_encode(const float* __restrict__ x,
                                                  const float* __restrict__ Wenc,
                                                  const float* __restrict__ benc,
                                                  const float* __restrict__ bdec,
                                                  float* __restrict__ feats) {
  __shared__ __attribute__((aligned(32))) __bf16 As[BM * BK]; // [row][k]
  __shared__ __attribute__((aligned(32))) __bf16 Bs[BN * BK]; // [col][k]

  const int tid  = threadIdx.x;
  const int lane = tid & 31;
  const int w    = tid >> 5;   // 0..7
  const int wm   = w >> 1;     // 0..3 : 32-row band
  const int wn   = w & 1;      // 0..1 : 64-col band

  const int tileM = blockIdx.x * BM;   // 16 tiles (fastest varying)
  const int tileN = blockIdx.y * BN;   // 512 tiles

  v8f acc[2][4];
#pragma unroll
  for (int m = 0; m < 2; ++m)
#pragma unroll
    for (int n = 0; n < 4; ++n) acc[m][n] = (v8f){};

  // loader assignments
  const int arow  = tid >> 1;          // 0..127
  const int akseg = (tid & 1) * 16;    // 0 or 16
  const int bcol  = tid & 127;         // 0..127
  const int bkseg = (tid >> 7) * 16;   // 0 or 16

  const float* aBase = x + (size_t)(tileM + arow) * DIN + akseg;
  const float* dBase = bdec + akseg;
  const float* bBase = Wenc + (size_t)bkseg * DSAE + tileN + bcol;

  __bf16 ast[16], bst[16];
  auto stage = [&](int k0) {
#pragma unroll
    for (int i = 0; i < 16; i += 4) {
      v4f v = *(const v4f*)(aBase + k0 + i);
      v4f d = *(const v4f*)(dBase + k0 + i);
      ast[i + 0] = f2bf(v.x - d.x);
      ast[i + 1] = f2bf(v.y - d.y);
      ast[i + 2] = f2bf(v.z - d.z);
      ast[i + 3] = f2bf(v.w - d.w);
    }
    const float* bp = bBase + (size_t)k0 * DSAE;
#pragma unroll
    for (int i = 0; i < 16; ++i) bst[i] = f2bf(bp[(size_t)i * DSAE]);
  };

  stage(0);  // prologue

  for (int k0 = 0; k0 < DIN; k0 += BK) {
    // commit staged tile to LDS
    {
      v8bf lo, hi;
#pragma unroll
      for (int i = 0; i < 8; ++i) { lo[i] = ast[i]; hi[i] = ast[i + 8]; }
      *(v8bf*)&As[arow * BK + akseg]     = lo;
      *(v8bf*)&As[arow * BK + akseg + 8] = hi;
#pragma unroll
      for (int i = 0; i < 8; ++i) { lo[i] = bst[i]; hi[i] = bst[i + 8]; }
      *(v8bf*)&Bs[bcol * BK + bkseg]     = lo;
      *(v8bf*)&Bs[bcol * BK + bkseg + 8] = hi;
    }
    __syncthreads();

    // prefetch next tile into registers while WMMAs run below
    if (k0 + BK < DIN) stage(k0 + BK);

    // fragments
    v16bf afrag[2], bfrag[4];
#pragma unroll
    for (int m = 0; m < 2; ++m) {
      const int row = wm * 32 + m * 16 + (lane & 15);
      const int kb  = ((lane >> 4) & 1) * 8;
      v8bf lo = *(const v8bf*)&As[row * BK + kb];
      v8bf hi = *(const v8bf*)&As[row * BK + kb + 16];
#pragma unroll
      for (int i = 0; i < 8; ++i) { afrag[m][i] = lo[i]; afrag[m][i + 8] = hi[i]; }
    }
#pragma unroll
    for (int n = 0; n < 4; ++n) {
      const int col = wn * 64 + n * 16 + (lane & 15);
      const int kb  = ((lane >> 4) & 1) * 8;
      v8bf lo = *(const v8bf*)&Bs[col * BK + kb];
      v8bf hi = *(const v8bf*)&Bs[col * BK + kb + 16];
#pragma unroll
      for (int i = 0; i < 8; ++i) { bfrag[n][i] = lo[i]; bfrag[n][i + 8] = hi[i]; }
    }
#pragma unroll
    for (int m = 0; m < 2; ++m)
#pragma unroll
      for (int n = 0; n < 4; ++n)
        acc[m][n] = __builtin_amdgcn_wmma_f32_16x16x32_bf16(
            false, afrag[m], false, bfrag[n], (short)0, acc[m][n], false, false);

    __syncthreads();
  }

  // epilogue: + b_enc, relu, store fp32
#pragma unroll
  for (int m = 0; m < 2; ++m) {
#pragma unroll
    for (int n = 0; n < 4; ++n) {
      const int col = tileN + wn * 64 + n * 16 + (lane & 15);
      const float be = benc[col];
      const int rowbase = tileM + wm * 32 + m * 16 + ((lane >> 4) & 1) * 8;
      float* op = feats + (size_t)rowbase * DSAE + col;
#pragma unroll
      for (int r = 0; r < 8; ++r) {
        float v = acc[m][n][r] + be;
        op[(size_t)r * DSAE] = v > 0.0f ? v : 0.0f;
      }
    }
  }
}

// ---------------------------------------------------------------------------
// Kernel B: per-token top-k -> sparsify feats in place -> sparse decode.
// grid = TOK blocks, block = 256
// ---------------------------------------------------------------------------
__global__ __launch_bounds__(256) void sae_topk_decode(const float* __restrict__ x,
                                                       const float* __restrict__ Wdec,
                                                       const float* __restrict__ bdec,
                                                       float* __restrict__ feats,
                                                       float* __restrict__ sae_out) {
  __shared__ float cand[256 * 32];  // per-thread sorted(asc) top-32
  __shared__ float red[256];
  __shared__ float topv[KTOP];
  __shared__ int   topi[KTOP];
  __shared__ int   taken;
  __shared__ int   cnt;

  const int tid = threadIdx.x;
  const int tok = blockIdx.x;
  float* row = feats + (size_t)tok * DSAE;

  if (tid < KTOP) { topv[tid] = 0.0f; topi[tid] = 0; }

  // pass 1: per-thread top-32 (insertion sort kept ascending in LDS)
  float* my = &cand[tid * 32];
#pragma unroll
  for (int i = 0; i < 32; ++i) my[i] = -1e30f;
  for (int i = tid; i < DSAE; i += 256) {
    float v = row[i];
    if (v > my[0]) {
      int p = 0;
      while (p < 31 && my[p + 1] < v) { my[p] = my[p + 1]; ++p; }
      my[p] = v;
    }
  }
  __syncthreads();

  // 32 rounds of block-wide max extraction; T = 32nd largest overall
  float T = -1e30f;
  for (int r = 0; r < KTOP; ++r) {
    float lm = my[0];
#pragma unroll
    for (int i = 1; i < 32; ++i) lm = fmaxf(lm, my[i]);
    red[tid] = lm;
    __syncthreads();
    for (int s = 128; s > 0; s >>= 1) {
      if (tid < s) red[tid] = fmaxf(red[tid], red[tid + s]);
      __syncthreads();
    }
    const float gm = red[0];
    __syncthreads();
    if (tid == 0) taken = 0;
    __syncthreads();
    bool have = false;
#pragma unroll
    for (int i = 0; i < 32; ++i) have = have || (my[i] == gm);
    if (have && atomicCAS(&taken, 0, 1) == 0) {
      for (int i = 31; i >= 0; --i) {
        if (my[i] == gm) { my[i] = -1e30f; break; }
      }
    }
    T = gm;
    __syncthreads();
  }

  if (tid == 0) cnt = 0;
  __syncthreads();

  // pass 2: sparsify in place + collect (val, idx) of survivors
  for (int i = tid; i < DSAE; i += 256) {
    float v = row[i];
    bool keep = false;
    if (v >= T) {
      int slot = atomicAdd(&cnt, 1);
      if (slot < KTOP) { topv[slot] = v; topi[slot] = i; keep = true; }
    }
    row[i] = keep ? v : 0.0f;
  }
  __syncthreads();

  // sparse decode: sae_out[tok,:] = sum_j topv[j] * Wdec[topi[j],:] + bdec
#pragma unroll
  for (int c0 = 0; c0 < DIN / 256; ++c0) {
    const int col = tid + c0 * 256;
    float a = bdec[col];
    for (int j = 0; j < KTOP; ++j)
      a += topv[j] * Wdec[(size_t)topi[j] * DIN + col];
    sae_out[(size_t)tok * DIN + col] = a;
  }
}

// ---------------------------------------------------------------------------
// Kernel C: partial per-column stats over a row-group (atomic-free).
// part layout in ws: [3][NROWG][DIN] = e, s, s2
// grid = (DIN/256, NROWG), block = 256
// ---------------------------------------------------------------------------
__global__ __launch_bounds__(256) void sae_colstats(const float* __restrict__ x,
                                                    const float* __restrict__ sae_out,
                                                    float* __restrict__ part) {
  const int col = blockIdx.x * 256 + threadIdx.x;
  const int g   = blockIdx.y;
  const int n0  = g * (TOK / NROWG);
  float e = 0.0f, s = 0.0f, s2 = 0.0f;
  for (int n = n0; n < n0 + TOK / NROWG; ++n) {
    const float xv = x[(size_t)n * DIN + col];
    const float ov = sae_out[(size_t)n * DIN + col];
    const float d = ov - xv;
    e += d * d;
    s += xv;
    s2 += xv * xv;
  }
  part[(size_t)0 * NROWG * DIN + (size_t)g * DIN + col] = e;
  part[(size_t)1 * NROWG * DIN + (size_t)g * DIN + col] = s;
  part[(size_t)2 * NROWG * DIN + (size_t)g * DIN + col] = s2;
}

// ---------------------------------------------------------------------------
// Kernel D: fvu = mean_col( l2 / (sum x^2 - (sum x)^2/N) )
// ---------------------------------------------------------------------------
__global__ __launch_bounds__(256) void sae_fvu(const float* __restrict__ part,
                                               float* __restrict__ fvu) {
  __shared__ float red[256];
  float acc = 0.0f;
  for (int c = threadIdx.x; c < DIN; c += 256) {
    float e = 0.0f, s = 0.0f, s2 = 0.0f;
    for (int g = 0; g < NROWG; ++g) {
      e  += part[(size_t)0 * NROWG * DIN + (size_t)g * DIN + c];
      s  += part[(size_t)1 * NROWG * DIN + (size_t)g * DIN + c];
      s2 += part[(size_t)2 * NROWG * DIN + (size_t)g * DIN + c];
    }
    const float var = s2 - s * s * (1.0f / (float)TOK);
    acc += e / var;
  }
  red[threadIdx.x] = acc;
  __syncthreads();
  for (int off = 128; off > 0; off >>= 1) {
    if (threadIdx.x < off) red[threadIdx.x] += red[threadIdx.x + off];
    __syncthreads();
  }
  if (threadIdx.x == 0) *fvu = red[0] * (1.0f / (float)DIN);
}

extern "C" void kernel_launch(void* const* d_in, const int* in_sizes, int n_in,
                              void* d_out, int out_size, void* d_ws, size_t ws_size,
                              hipStream_t stream) {
  const float* x    = (const float*)d_in[0];
  const float* Wenc = (const float*)d_in[1];
  const float* benc = (const float*)d_in[2];
  const float* Wdec = (const float*)d_in[3];
  const float* bdec = (const float*)d_in[4];
  // d_in[5] = k (==32, compile-time KTOP)

  float* sae_out = (float*)d_out;                       // [TOK, DIN]
  float* feats   = sae_out + (size_t)TOK * DIN;         // [TOK, DSAE]
  float* fvu     = feats + (size_t)TOK * DSAE;          // scalar

  float* part = (float*)d_ws;                           // [3][NROWG][DIN]

  dim3 gA(TOK / BM, DSAE / BN);                         // (16, 512) M fastest
  sae_encode<<<gA, 256, 0, stream>>>(x, Wenc, benc, bdec, feats);
  sae_topk_decode<<<TOK, 256, 0, stream>>>(x, Wdec, bdec, feats, sae_out);
  sae_colstats<<<dim3(DIN / 256, NROWG), 256, 0, stream>>>(x, sae_out, part);
  sae_fvu<<<1, 256, 0, stream>>>(part, fvu);
}
